// GraphNeuralNetwork_57337813401889
// MI455X (gfx1250) — matrix operand order repaired and checked by reading, not compiled, not run on hardware
//
#include <hip/hip_runtime.h>

// ---------------------------------------------------------------------------
// CDNA5 (gfx1250) GNN: fused edge-MLP + segment-sum + node-update MLPs,
// all matmuls on v_wmma_f32_16x16x32_bf16 (wave32 WMMA).
// Round 2: batch all fragment loads of a K-step before the WMMA chain so the
// scheduler can overlap ds_load latency with preceding WMMAs instead of
// emitting s_wait_dscnt 0 before every matrix op.
// ---------------------------------------------------------------------------

typedef __bf16 bf16;
typedef __attribute__((ext_vector_type(16))) __bf16 v16bf;
typedef __attribute__((ext_vector_type(8)))  float  v8f;
typedef __attribute__((ext_vector_type(4)))  unsigned int v4u;

union FragU { v16bf bf; v4u q[2]; };

// A-matrix fragment (16x32 bf16), source row-major [16+][stride] in LDS.
// Lane layout (ISA 7.12.2): lane L holds row m=L%16; K elements
// half*8..half*8+7 (VGPR0-3) and 16+half*8..+7 (VGPR4-7), half=L/16.
__device__ __forceinline__ v16bf load_frag_a(const bf16* base, int stride, int lane) {
  const int m = lane & 15, half = lane >> 4;
  const bf16* p = base + m * stride + half * 8;
  FragU f;
  f.q[0] = *(const v4u*)p;
  f.q[1] = *(const v4u*)(p + 16);
  return f.bf;
}

// B-matrix fragment (32x16 bf16) from LDS holding W^T row-major [N][stride]:
// lane L holds column n=L%16, K = half*8..half*8+7 and 16+half*8..+7.
__device__ __forceinline__ v16bf load_frag_b(const bf16* base, int stride, int lane) {
  const int n = lane & 15, half = lane >> 4;
  const bf16* p = base + n * stride + half * 8;
  FragU f;
  f.q[0] = *(const v4u*)p;
  f.q[1] = *(const v4u*)(p + 16);
  return f.bf;
}

__device__ __forceinline__ v8f wmma_bf16(v16bf a, v16bf b, v8f c) {
  return __builtin_amdgcn_wmma_f32_16x16x32_bf16(false, a, false, b, (short)0, c,
                                                 false, false);
}

__device__ __forceinline__ unsigned int pack2bf(float a, float b) {
  union { __bf16 h[2]; unsigned int u; } x;
  x.h[0] = (__bf16)a; x.h[1] = (__bf16)b;
  return x.u;
}

// ---------------------------------------------------------------------------
// Edge kernel: for a 64-edge tile, compute both m_v and m_c through the
// shared messenger MLP and atomically accumulate into agg_v / agg_c.
//   layer1: X[64 x 160] @ W1^T-staged  -> ReLU -> H1[64 x 128] (bf16 in LDS)
//   layer2: H1 @ W2^T-staged -> M[64 x 64] -> atomicAdd scatter
// K chunks (32 each): side 0: [vs,vs,cs,cs,ef] ; side 1: [cs,cs,vs,vs,ef]
// ---------------------------------------------------------------------------
__global__ void __launch_bounds__(256)
gnn_edge_kernel(const float* __restrict__ vfeat, const float* __restrict__ cfeat,
                const float* __restrict__ efeat,
                const int* __restrict__ edge_var, const int* __restrict__ edge_cons,
                const bf16* __restrict__ w1t,  // [128][160] bf16 (transposed, K-padded)
                const bf16* __restrict__ w2t,  // [64][128]  bf16 (transposed)
                const float* __restrict__ b1, const float* __restrict__ b2,
                float* __restrict__ aggv, float* __restrict__ aggc, int E)
{
  extern __shared__ char smem_raw[];
  bf16* s_w1t = (bf16*)smem_raw;            // 128*160
  bf16* s_w2t = s_w1t + 128 * 160;          // 64*128
  bf16* s_vs  = s_w2t + 64 * 128;           // 64*64
  bf16* s_cs  = s_vs + 64 * 64;             // 64*64
  bf16* s_ef  = s_cs + 64 * 64;             // 64*32 (8 real cols, zero-padded)
  bf16* s_h1  = s_ef + 64 * 32;             // 64*128
  int*  s_ev  = (int*)(s_h1 + 64 * 128);    // 64
  int*  s_ec  = s_ev + 64;                  // 64

  const int tid  = threadIdx.x;
  const int lane = tid & 31;
  const int wave = tid >> 5;
  const int l16  = lane & 15;
  const int half = lane >> 4;
  const int mw   = wave & 3;   // 4-way M split (16 rows each)
  const int nw   = wave >> 2;  // 2-way N split

  // Stage weights into LDS once per (persistent) block.
  {
    const v4u* s1 = (const v4u*)w1t; v4u* d1 = (v4u*)s_w1t;
    for (int i = tid; i < (128 * 160) / 8; i += 256) d1[i] = s1[i];
    const v4u* s2 = (const v4u*)w2t; v4u* d2 = (v4u*)s_w2t;
    for (int i = tid; i < (64 * 128) / 8; i += 256) d2[i] = s2[i];
  }

  // Per-lane bias values for the output columns this lane owns.
  float bias1[4], bias2[2];
#pragma unroll
  for (int a = 0; a < 4; ++a) bias1[a] = b1[64 * nw + 16 * a + l16];
#pragma unroll
  for (int a = 0; a < 2; ++a) bias2[a] = b2[32 * nw + 16 * a + l16];

  const int numTiles = (E + 63) >> 6;
  for (int tile = blockIdx.x; tile < numTiles; tile += gridDim.x) {
    const int ebase = tile << 6;
    __syncthreads();  // previous tile fully consumed (also covers weight staging)

    if (tid < 64) {
      const int e = ebase + tid;
      s_ev[tid] = (e < E) ? edge_var[e]  : 0;
      s_ec[tid] = (e < E) ? edge_cons[e] : 0;
    }
    __syncthreads();

    // Gather var / cons endpoint features -> bf16 tiles (float2 granularity).
    for (int i = tid; i < 64 * 32; i += 256) {
      const int r = i >> 5, cp = i & 31;
      const int e = ebase + r;
      unsigned int pv = 0, pc = 0;
      if (e < E) {
        const float2 v = ((const float2*)(vfeat + (size_t)s_ev[r] * 64))[cp];
        const float2 c = ((const float2*)(cfeat + (size_t)s_ec[r] * 64))[cp];
        pv = pack2bf(v.x, v.y);
        pc = pack2bf(c.x, c.y);
      }
      ((unsigned int*)s_vs)[i] = pv;
      ((unsigned int*)s_cs)[i] = pc;
    }
    // Edge features (8 real cols -> 32 padded cols).
    for (int i = tid; i < 64 * 16; i += 256) {
      const int r = i >> 4, cp = i & 15;
      const int e = ebase + r;
      unsigned int p = 0;
      if (e < E && cp < 4) {
        const float2 v = ((const float2*)(efeat + (size_t)e * 8))[cp];
        p = pack2bf(v.x, v.y);
      }
      ((unsigned int*)s_ef)[i] = p;
    }
    __syncthreads();

    for (int side = 0; side < 2; ++side) {
      const bf16* Xa = side ? s_cs : s_vs;  // K chunks 0-1
      const bf16* Xb = side ? s_vs : s_cs;  // K chunks 2-3

      // ---- layer 1: [64 x 160] x [160 x 128] ----
      v8f acc[4];
#pragma unroll
      for (int a = 0; a < 4; ++a) {
        v8f t;
#pragma unroll
        for (int v = 0; v < 8; ++v) t[v] = bias1[a];
        acc[a] = t;
      }
#pragma unroll
      for (int kc = 0; kc < 5; ++kc) {
        const bf16* abase; int astride;
        if (kc < 2)      { abase = Xa + (16 * mw) * 64 + kc * 32;       astride = 64; }
        else if (kc < 4) { abase = Xb + (16 * mw) * 64 + (kc - 2) * 32; astride = 64; }
        else             { abase = s_ef + (16 * mw) * 32;               astride = 32; }
        // Batch all fragment loads of this K-step before the WMMA chain.
        const v16bf af = load_frag_a(abase, astride, lane);
        v16bf bfr[4];
#pragma unroll
        for (int a = 0; a < 4; ++a)
          bfr[a] = load_frag_b(s_w1t + (64 * nw + 16 * a) * 160 + kc * 32, 160, lane);
#pragma unroll
        for (int a = 0; a < 4; ++a)
          acc[a] = wmma_bf16(af, bfr[a], acc[a]);
      }

      // ReLU, convert, stage H1 (barrier protects previous side's layer-2 reads).
      __syncthreads();
#pragma unroll
      for (int a = 0; a < 4; ++a) {
#pragma unroll
        for (int v = 0; v < 8; ++v) {
          float x = acc[a][v];
          x = x > 0.0f ? x : 0.0f;
          s_h1[(16 * mw + 8 * half + v) * 128 + 64 * nw + 16 * a + l16] = (bf16)x;
        }
      }
      __syncthreads();

      // ---- layer 2: [64 x 128] x [128 x 64] ----
      v8f acc2[2];
#pragma unroll
      for (int a = 0; a < 2; ++a) {
        v8f t;
#pragma unroll
        for (int v = 0; v < 8; ++v) t[v] = bias2[a];
        acc2[a] = t;
      }
#pragma unroll
      for (int kc = 0; kc < 4; ++kc) {
        const v16bf af = load_frag_a(s_h1 + (16 * mw) * 128 + kc * 32, 128, lane);
        v16bf bfr[2];
#pragma unroll
        for (int a = 0; a < 2; ++a)
          bfr[a] = load_frag_b(s_w2t + (32 * nw + 16 * a) * 128 + kc * 32, 128, lane);
#pragma unroll
        for (int a = 0; a < 2; ++a)
          acc2[a] = wmma_bf16(af, bfr[a], acc2[a]);
      }

      // ---- segment-sum scatter ----
      const int* sidx = side ? s_ec : s_ev;
      float* agg = side ? aggc : aggv;
#pragma unroll
      for (int a = 0; a < 2; ++a) {
        const int col = 32 * nw + 16 * a + l16;
#pragma unroll
        for (int v = 0; v < 8; ++v) {
          const int r = 16 * mw + 8 * half + v;
          if (ebase + r < E)
            atomicAdd(agg + (size_t)sidx[r] * 64 + col, acc2[a][v]);
        }
      }
    }
  }
}

// ---------------------------------------------------------------------------
// Node update: out = MLP(concat[feat, agg]) ; X = [64 x 128], W1 [128x128],
// W2 [128x64]. Same WMMA tiling as the edge kernel.
// ---------------------------------------------------------------------------
__global__ void __launch_bounds__(256)
gnn_node_kernel(const float* __restrict__ feat, const float* __restrict__ agg,
                const bf16* __restrict__ w1t,  // [128][128] transposed
                const bf16* __restrict__ w2t,  // [64][128] transposed
                const float* __restrict__ b1, const float* __restrict__ b2,
                float* __restrict__ out, int N)
{
  extern __shared__ char smem_raw[];
  bf16* s_w1t = (bf16*)smem_raw;          // 128*128
  bf16* s_w2t = s_w1t + 128 * 128;        // 64*128
  bf16* s_x0  = s_w2t + 64 * 128;         // 64*64  (feat)
  bf16* s_x1  = s_x0 + 64 * 64;           // 64*64  (agg)
  bf16* s_h1  = s_x1 + 64 * 64;           // 64*128

  const int tid  = threadIdx.x;
  const int lane = tid & 31;
  const int wave = tid >> 5;
  const int l16  = lane & 15;
  const int half = lane >> 4;
  const int mw   = wave & 3;
  const int nw   = wave >> 2;

  {
    const v4u* s1 = (const v4u*)w1t; v4u* d1 = (v4u*)s_w1t;
    for (int i = tid; i < (128 * 128) / 8; i += 256) d1[i] = s1[i];
    const v4u* s2 = (const v4u*)w2t; v4u* d2 = (v4u*)s_w2t;
    for (int i = tid; i < (64 * 128) / 8; i += 256) d2[i] = s2[i];
  }

  float bias1[4], bias2[2];
#pragma unroll
  for (int a = 0; a < 4; ++a) bias1[a] = b1[64 * nw + 16 * a + l16];
#pragma unroll
  for (int a = 0; a < 2; ++a) bias2[a] = b2[32 * nw + 16 * a + l16];

  const int numTiles = (N + 63) >> 6;
  for (int tile = blockIdx.x; tile < numTiles; tile += gridDim.x) {
    const int rbase = tile << 6;
    __syncthreads();

    for (int i = tid; i < 64 * 32; i += 256) {
      const int r = i >> 5, cp = i & 31;
      const int row = rbase + r;
      unsigned int p0 = 0, p1 = 0;
      if (row < N) {
        const float2 a = ((const float2*)(feat + (size_t)row * 64))[cp];
        const float2 b = ((const float2*)(agg  + (size_t)row * 64))[cp];
        p0 = pack2bf(a.x, a.y);
        p1 = pack2bf(b.x, b.y);
      }
      ((unsigned int*)s_x0)[i] = p0;
      ((unsigned int*)s_x1)[i] = p1;
    }
    __syncthreads();

    // ---- layer 1 ----
    v8f acc[4];
#pragma unroll
    for (int a = 0; a < 4; ++a) {
      v8f t;
#pragma unroll
      for (int v = 0; v < 8; ++v) t[v] = bias1[a];
      acc[a] = t;
    }
#pragma unroll
    for (int kc = 0; kc < 4; ++kc) {
      const bf16* abase = (kc < 2) ? (s_x0 + (16 * mw) * 64 + kc * 32)
                                   : (s_x1 + (16 * mw) * 64 + (kc - 2) * 32);
      const v16bf af = load_frag_a(abase, 64, lane);
      v16bf bfr[4];
#pragma unroll
      for (int a = 0; a < 4; ++a)
        bfr[a] = load_frag_b(s_w1t + (64 * nw + 16 * a) * 128 + kc * 32, 128, lane);
#pragma unroll
      for (int a = 0; a < 4; ++a)
        acc[a] = wmma_bf16(af, bfr[a], acc[a]);
    }

    __syncthreads();
#pragma unroll
    for (int a = 0; a < 4; ++a) {
#pragma unroll
      for (int v = 0; v < 8; ++v) {
        float x = acc[a][v];
        x = x > 0.0f ? x : 0.0f;
        s_h1[(16 * mw + 8 * half + v) * 128 + 64 * nw + 16 * a + l16] = (bf16)x;
      }
    }
    __syncthreads();

    // ---- layer 2 ----
    v8f acc2[2];
#pragma unroll
    for (int a = 0; a < 2; ++a) {
      v8f t;
#pragma unroll
      for (int v = 0; v < 8; ++v) t[v] = bias2[a];
      acc2[a] = t;
    }
#pragma unroll
    for (int kc = 0; kc < 4; ++kc) {
      const v16bf af = load_frag_a(s_h1 + (16 * mw) * 128 + kc * 32, 128, lane);
      v16bf bfr[2];
#pragma unroll
      for (int a = 0; a < 2; ++a)
        bfr[a] = load_frag_b(s_w2t + (32 * nw + 16 * a) * 128 + kc * 32, 128, lane);
#pragma unroll
      for (int a = 0; a < 2; ++a)
        acc2[a] = wmma_bf16(af, bfr[a], acc2[a]);
    }

#pragma unroll
    for (int a = 0; a < 2; ++a) {
      const int col = 32 * nw + 16 * a + l16;
#pragma unroll
      for (int v = 0; v < 8; ++v) {
        const int r = 16 * mw + 8 * half + v;
        if (rbase + r < N) out[(size_t)(rbase + r) * 64 + col] = acc2[a][v];
      }
    }
  }
}

// ---------------------------------------------------------------------------
// Utility kernels
// ---------------------------------------------------------------------------
__global__ void convert_wT(const float* __restrict__ src, bf16* __restrict__ dst,
                           int K, int N, int Kpad)
{
  const int idx = blockIdx.x * blockDim.x + threadIdx.x;
  if (idx >= N * Kpad) return;
  const int n = idx / Kpad;
  const int k = idx - n * Kpad;
  const float v = (k < K) ? src[(size_t)k * N + n] : 0.0f;
  dst[idx] = (bf16)v;
}

__global__ void zero_f32(float* __restrict__ p, int n)
{
  const int stride = gridDim.x * blockDim.x;
  for (int i = blockIdx.x * blockDim.x + threadIdx.x; i < n; i += stride) p[i] = 0.0f;
}

// ---------------------------------------------------------------------------
// Host orchestration
// ---------------------------------------------------------------------------
extern "C" void kernel_launch(void* const* d_in, const int* in_sizes, int n_in,
                              void* d_out, int out_size, void* d_ws, size_t ws_size,
                              hipStream_t stream)
{
  const float* var_feat  = (const float*)d_in[0];
  const float* cons_feat = (const float*)d_in[1];
  const float* edge_feat = (const float*)d_in[2];
  const int*   edge_var  = (const int*)d_in[3];
  const int*   edge_cons = (const int*)d_in[4];
  const float* w_msg1  = (const float*)d_in[5];
  const float* b_msg1  = (const float*)d_in[6];
  const float* w_msg2  = (const float*)d_in[7];
  const float* b_msg2  = (const float*)d_in[8];
  const float* w_vupd1 = (const float*)d_in[9];
  const float* b_vupd1 = (const float*)d_in[10];
  const float* w_vupd2 = (const float*)d_in[11];
  const float* b_vupd2 = (const float*)d_in[12];
  const float* w_cupd1 = (const float*)d_in[13];
  const float* b_cupd1 = (const float*)d_in[14];
  const float* w_cupd2 = (const float*)d_in[15];
  const float* b_cupd2 = (const float*)d_in[16];

  const int F = 64, H = 128;
  const int MSG_IN = 2 * F + 8;     // 136
  const int MSG_KP = 160;           // padded to 5 x 32
  const int NV = in_sizes[0] / F;
  const int NC = in_sizes[1] / F;
  const int E  = in_sizes[3];
  const int ITERS = in_sizes[6] / H;  // b_msg1 is [ITERS, H]

  // ---- workspace carve-up (256B aligned) ----
  char* ws = (char*)d_ws;
  size_t off = 0;
  auto wsalloc = [&](size_t bytes) -> void* {
    void* p = ws + off;
    off += (bytes + 255) & ~(size_t)255;
    return p;
  };
  float* aggv = (float*)wsalloc((size_t)NV * F * sizeof(float));
  float* aggc = (float*)wsalloc((size_t)NC * F * sizeof(float));
  float* tmpv = (float*)wsalloc((size_t)NV * F * sizeof(float));
  float* tmpc = (float*)wsalloc((size_t)NC * F * sizeof(float));
  bf16* msg1T  = (bf16*)wsalloc((size_t)ITERS * H * MSG_KP * sizeof(bf16));
  bf16* msg2T  = (bf16*)wsalloc((size_t)ITERS * F * H * sizeof(bf16));
  bf16* vupd1T = (bf16*)wsalloc((size_t)ITERS * H * H * sizeof(bf16));
  bf16* vupd2T = (bf16*)wsalloc((size_t)ITERS * F * H * sizeof(bf16));
  bf16* cupd1T = (bf16*)wsalloc((size_t)ITERS * H * H * sizeof(bf16));
  bf16* cupd2T = (bf16*)wsalloc((size_t)ITERS * F * H * sizeof(bf16));

  // ---- convert weights to transposed bf16 (once per launch) ----
  auto cv = [&](const float* src, bf16* dst, int K, int N, int Kpad) {
    const int total = N * Kpad;
    convert_wT<<<(total + 255) / 256, 256, 0, stream>>>(src, dst, K, N, Kpad);
  };
  for (int i = 0; i < ITERS; ++i) {
    cv(w_msg1  + (size_t)i * MSG_IN * H, msg1T  + (size_t)i * H * MSG_KP, MSG_IN, H, MSG_KP);
    cv(w_msg2  + (size_t)i * H * F,      msg2T  + (size_t)i * F * H,      H,      F, H);
    cv(w_vupd1 + (size_t)i * H * H,      vupd1T + (size_t)i * H * H,      H,      H, H);
    cv(w_vupd2 + (size_t)i * H * F,      vupd2T + (size_t)i * F * H,      H,      F, H);
    cv(w_cupd1 + (size_t)i * H * H,      cupd1T + (size_t)i * H * H,      H,      H, H);
    cv(w_cupd2 + (size_t)i * H * F,      cupd2T + (size_t)i * F * H,      H,      F, H);
  }

  const size_t edge_smem =
      (size_t)(128 * 160 + 64 * 128 + 64 * 64 + 64 * 64 + 64 * 32 + 64 * 128) * 2 + 128 * 4;
  const size_t node_smem =
      (size_t)(128 * 128 + 64 * 128 + 64 * 64 + 64 * 64 + 64 * 128) * 2;

  const float* vin = var_feat;
  const float* cin = cons_feat;
  for (int i = 0; i < ITERS; ++i) {
    zero_f32<<<1024, 256, 0, stream>>>(aggv, NV * F);
    zero_f32<<<1024, 256, 0, stream>>>(aggc, NC * F);

    gnn_edge_kernel<<<1280, 256, edge_smem, stream>>>(
        vin, cin, edge_feat, edge_var, edge_cons,
        msg1T + (size_t)i * H * MSG_KP, msg2T + (size_t)i * F * H,
        b_msg1 + (size_t)i * H, b_msg2 + (size_t)i * F,
        aggv, aggc, E);

    float* vout = (i == ITERS - 1) ? (float*)d_out : tmpv;
    float* cout = (i == ITERS - 1) ? ((float*)d_out + (size_t)NV * F) : tmpc;

    const int vtiles = (NV + 63) / 64;
    const int ctiles = (NC + 63) / 64;
    gnn_node_kernel<<<vtiles, 256, node_smem, stream>>>(
        vin, aggv, vupd1T + (size_t)i * H * H, vupd2T + (size_t)i * F * H,
        b_vupd1 + (size_t)i * H, b_vupd2 + (size_t)i * F, vout, NV);
    gnn_node_kernel<<<ctiles, 256, node_smem, stream>>>(
        cin, aggc, cupd1T + (size_t)i * H * H, cupd2T + (size_t)i * F * H,
        b_cupd1 + (size_t)i * H, b_cupd2 + (size_t)i * F, cout, NC);

    vin = vout;
    cin = cout;
  }
}
